// RNN_44212393345116
// MI455X (gfx1250) — compile-verified, matrix operand order checked by reading, not run on hardware
//
#include <hip/hip_runtime.h>
#include <hip/hip_bf16.h>

// ---------------------------------------------------------------------------
// 2-layer LSTM for MI455X (gfx1250, wave32, WMMA).
//   z = [x_t | h_{t-1}] @ W + b   -> i,j,f,o gates -> c,h update
// Per-step GEMM: [64 x 2048] @ [2048 x 4096] in bf16 WMMA, fp32 accumulate.
// W is pre-transposed to Wt[4096][2048] bf16 so both A and B fragments are
// contiguous 16-byte loads in the wave32 WMMA lane layout.
// A (=[x|h], 64x2048 bf16, ~263KB padded) is staged into the 320KB WGP LDS
// with GLOBAL_LOAD_ASYNC_TO_LDS_B128 (ASYNCcnt-tracked global->LDS DMA).
// B fragments are double-buffered in registers (prefetch distance = 1 chunk).
// ---------------------------------------------------------------------------

typedef __attribute__((ext_vector_type(16))) __bf16 v16bf;
typedef __attribute__((ext_vector_type(8)))  float  v8f;

#define T_STEPS 512
#define BATCH   64
#define HID     1024
#define KTOT    2048            // D + H
#define NGATE   4096            // 4 * HID
#define LDS_ROW 2056            // 2048 + 8 bf16 pad (16B) -> rotates LDS banks by 4/row
#define LDS_ROW_B (LDS_ROW * 2) // 4112 bytes per row
#define SMEM_BYTES (64 * LDS_ROW_B)     // 263168 B of dynamic LDS (<= 320KB/WGP)

union Frag {
    v16bf v;
    uint4 u[2];
};

__device__ __forceinline__ float sigmoidf_(float x) {
    return 1.0f / (1.0f + __expf(-x));
}

// Per-lane async global->LDS 16-byte copy (gfx1250, ASYNCcnt-tracked).
__device__ __forceinline__ void async_copy_b128(uint32_t lds_addr, uint64_t gaddr) {
    asm volatile("global_load_async_to_lds_b128 %0, %1, off"
                 :: "v"(lds_addr), "v"(gaddr) : "memory");
}

// ---------------------------------------------------------------------------
// One LSTM time step. Grid: 16 blocks x 256 threads (8 waves).
// Wave w of block g owns hidden tile ht = 4*g + (w>>1) (16 hidden units) and
// row half mbase = 32*(w&1). It computes 4 gate tiles x 2 M-tiles = 8 WMMA
// accumulators, K loop = 2048 / 32.
// ---------------------------------------------------------------------------
__global__ __launch_bounds__(256, 1)
void lstm_step_kernel(const __bf16* __restrict__ xt,      // [64][1024] input at t (bf16)
                      const __bf16* __restrict__ hprev,   // [64][1024] h_{t-1} (bf16)
                      const __bf16* __restrict__ wt,      // [4096][2048] W^T (bf16)
                      const float*  __restrict__ bias,    // [4096] fp32 (i,j,f,o)
                      float*        __restrict__ cstate,  // [64][1024] fp32 in/out
                      __bf16*       __restrict__ hout_bf, // [64][1024] bf16 out
                      float*        __restrict__ hout_f,  // [64][1024] fp32 out or null
                      int t0)                             // 1 if t == 0 (c = h = 0)
{
    extern __shared__ __bf16 smem[];
    uint4* s4 = reinterpret_cast<uint4*>(smem);           // row stride = 257 uint4

    const int tid = threadIdx.x;

    // ---- Stage A = [x_t | h_prev] into LDS via async DMA ------------------
    // Low 32 bits of a flat LDS pointer are the wave-relative LDS address
    // (ISA 10.2); no static LDS in this kernel.
    const uint32_t ldsbase = (uint32_t)(uintptr_t)smem;

    for (int idx = tid; idx < 64 * 128; idx += 256) {     // x half: 64 rows x 128 x 16B
        const int row = idx >> 7, ch = idx & 127;
        async_copy_b128(ldsbase + row * LDS_ROW_B + ch * 16,
                        (uint64_t)(uintptr_t)(xt + row * HID + ch * 8));
    }
    if (t0) {                                             // h half = zeros at t == 0
        const uint4 z = make_uint4(0u, 0u, 0u, 0u);
        for (int idx = tid; idx < 64 * 128; idx += 256) {
            const int row = idx >> 7, ch = idx & 127;
            s4[row * 257 + 128 + ch] = z;
        }
    } else {                                              // h half via async DMA
        for (int idx = tid; idx < 64 * 128; idx += 256) {
            const int row = idx >> 7, ch = idx & 127;
            async_copy_b128(ldsbase + row * LDS_ROW_B + 2048 + ch * 16,
                            (uint64_t)(uintptr_t)(hprev + row * HID + ch * 8));
        }
    }
    asm volatile("s_wait_asynccnt 0x0" ::: "memory");     // retire the DMA
    __syncthreads();

    // ---- Wave work assignment ---------------------------------------------
    const int lane  = tid & 31;
    const int wave  = tid >> 5;
    const int ht    = blockIdx.x * 4 + (wave >> 1);       // hidden tile 0..63
    const int h0    = ht << 4;
    const int mbase = (wave & 1) << 5;                    // 0 or 32
    const int lrow  = lane & 15;                          // row-in-tile / col-in-tile
    const int lhi   = lane >> 4;                          // selects K sub-chunk

    const v8f vzero = {0.f, 0.f, 0.f, 0.f, 0.f, 0.f, 0.f, 0.f};
    v8f acc[4][2];
#pragma unroll
    for (int g = 0; g < 4; ++g)
#pragma unroll
        for (int m = 0; m < 2; ++m) acc[g][m] = vzero;

    // A fragments: lane L holds row (mbase + mt*16 + lrow), K chunks at
    // k + 8*lhi and k + 16 + 8*lhi (per the 16-bit 16x32 A layout).
    const uint4* sA0 = s4 + (mbase + lrow) * 257;
    const uint4* sA1 = s4 + (mbase + 16 + lrow) * 257;

    // B fragments: column N = h0+lrow of gate g == row (g*HID + h0 + lrow) of Wt.
    const uint4* wg[4];
#pragma unroll
    for (int g = 0; g < 4; ++g)
        wg[g] = reinterpret_cast<const uint4*>(wt) +
                (size_t)(g * HID + h0 + lrow) * (KTOT / 8);

    constexpr int NK = KTOT / 8;                          // 256 uint4 per Wt row

    // ---- Software-pipelined K loop: 2 chunks (2x32 K) per iteration -------
    Frag bA[4], bB[4];
#pragma unroll
    for (int g = 0; g < 4; ++g) {                         // preload chunk 0
        bA[g].u[0] = wg[g][lhi];
        bA[g].u[1] = wg[g][lhi + 2];
    }

#pragma unroll 1
    for (int k8 = 0; k8 < NK; k8 += 8) {
        // Prefetch chunk k8+4 into the other buffer.
        const int kn = k8 + 4;
#pragma unroll
        for (int g = 0; g < 4; ++g) {
            bB[g].u[0] = wg[g][kn + lhi];
            bB[g].u[1] = wg[g][kn + lhi + 2];
        }
        // A fragments for both chunks (LDS, cheap).
        Frag a0, a1, c0_, c1_;
        a0.u[0]  = sA0[k8 + lhi];      a0.u[1]  = sA0[k8 + lhi + 2];
        a1.u[0]  = sA1[k8 + lhi];      a1.u[1]  = sA1[k8 + lhi + 2];
        c0_.u[0] = sA0[kn + lhi];      c0_.u[1] = sA0[kn + lhi + 2];
        c1_.u[0] = sA1[kn + lhi];      c1_.u[1] = sA1[kn + lhi + 2];

        // Chunk k8: consumes bA (loaded a full iteration ago).
#pragma unroll
        for (int g = 0; g < 4; ++g) {
            acc[g][0] = __builtin_amdgcn_wmma_f32_16x16x32_bf16(
                false, a0.v, false, bA[g].v, (short)0, acc[g][0], false, false);
            acc[g][1] = __builtin_amdgcn_wmma_f32_16x16x32_bf16(
                false, a1.v, false, bA[g].v, (short)0, acc[g][1], false, false);
        }

        // Prefetch chunk k8+8 (clamped: last prefetch is a dummy, never used).
        const int kf = (k8 + 8 < NK) ? (k8 + 8) : 0;
#pragma unroll
        for (int g = 0; g < 4; ++g) {
            bA[g].u[0] = wg[g][kf + lhi];
            bA[g].u[1] = wg[g][kf + lhi + 2];
        }

        // Chunk k8+4: consumes bB.
#pragma unroll
        for (int g = 0; g < 4; ++g) {
            acc[g][0] = __builtin_amdgcn_wmma_f32_16x16x32_bf16(
                false, c0_.v, false, bB[g].v, (short)0, acc[g][0], false, false);
            acc[g][1] = __builtin_amdgcn_wmma_f32_16x16x32_bf16(
                false, c1_.v, false, bB[g].v, (short)0, acc[g][1], false, false);
        }
    }

    // ---- Epilogue: gates. C-layout: VGPR r of a tile holds row
    // mbase + mt*16 + 8*lhi + r, column h0 + lrow. i/j/f/o for the same
    // (row, unit) live in the same lane slot across the 4 gate accumulators.
    const int   hcol = h0 + lrow;
    const float bi  = bias[hcol];
    const float bj  = bias[HID + hcol];
    const float bf_ = bias[2 * HID + hcol] + 1.0f;        // FORGET_BIAS
    const float bo  = bias[3 * HID + hcol];

#pragma unroll
    for (int m = 0; m < 2; ++m) {
#pragma unroll
        for (int r = 0; r < 8; ++r) {
            const int brow = mbase + m * 16 + lhi * 8 + r;
            const int off  = brow * HID + hcol;
            const float zi = acc[0][m][r] + bi;
            const float zj = acc[1][m][r] + bj;
            const float zf = acc[2][m][r] + bf_;
            const float zo = acc[3][m][r] + bo;
            const float cp = t0 ? 0.0f : cstate[off];
            const float cn = cp * sigmoidf_(zf) + sigmoidf_(zi) * tanhf(zj);
            const float hn = tanhf(cn) * sigmoidf_(zo);
            cstate[off]  = cn;
            hout_bf[off] = (__bf16)hn;
            if (hout_f) hout_f[off] = hn;
        }
    }
}

// ---------------------------------------------------------------------------
// One-time: W [2048][4096] f32 -> Wt [4096][2048] bf16 (LDS-tiled transpose).
// ---------------------------------------------------------------------------
__global__ __launch_bounds__(256)
void transpose_to_bf16(const float* __restrict__ w, __bf16* __restrict__ wt) {
    __shared__ float tile[32][33];
    const int kb = blockIdx.x * 32;
    const int nb = blockIdx.y * 32;
    const int tx = threadIdx.x & 31;
    const int ty = threadIdx.x >> 5;          // 0..7
    for (int i = ty; i < 32; i += 8)
        tile[i][tx] = w[(size_t)(kb + i) * NGATE + nb + tx];
    __syncthreads();
    for (int i = ty; i < 32; i += 8)
        wt[(size_t)(nb + i) * KTOT + kb + tx] = (__bf16)tile[tx][i];
}

// ---------------------------------------------------------------------------
// One-time: fp32 -> bf16 bulk convert.
// ---------------------------------------------------------------------------
__global__ __launch_bounds__(256)
void convert_f32_to_bf16(const float* __restrict__ in, __bf16* __restrict__ out, int n) {
    const int i = blockIdx.x * blockDim.x + threadIdx.x;
    if (i < n) out[i] = (__bf16)in[i];
}

// ---------------------------------------------------------------------------
extern "C" void kernel_launch(void* const* d_in, const int* in_sizes, int n_in,
                              void* d_out, int out_size, void* d_ws, size_t ws_size,
                              hipStream_t stream) {
    (void)in_sizes; (void)n_in; (void)out_size; (void)ws_size;

    const float* X  = (const float*)d_in[0];   // [512][64][1024]
    const float* W0 = (const float*)d_in[1];   // [2048][4096]
    const float* b0 = (const float*)d_in[2];   // [4096]
    const float* W1 = (const float*)d_in[3];   // [2048][4096]
    const float* b1 = (const float*)d_in[4];   // [4096]
    float* out = (float*)d_out;                // [512][64][1024]

    char* ws = (char*)d_ws;
    __bf16* Wt0 = (__bf16*)ws; ws += (size_t)NGATE * KTOT * 2;
    __bf16* Wt1 = (__bf16*)ws; ws += (size_t)NGATE * KTOT * 2;
    __bf16* Xbf = (__bf16*)ws; ws += (size_t)T_STEPS * BATCH * HID * 2;
    __bf16* H0  = (__bf16*)ws; ws += (size_t)T_STEPS * BATCH * HID * 2;
    __bf16* H1  = (__bf16*)ws; ws += (size_t)T_STEPS * BATCH * HID * 2;
    float*  c0  = (float*)ws;  ws += (size_t)BATCH * HID * 4;
    float*  c1  = (float*)ws;  ws += (size_t)BATCH * HID * 4;

    hipFuncSetAttribute((const void*)lstm_step_kernel,
                        hipFuncAttributeMaxDynamicSharedMemorySize, SMEM_BYTES);

    // One-time preprocessing (amortized over 1024 step kernels).
    dim3 tgrid(KTOT / 32, NGATE / 32);
    transpose_to_bf16<<<tgrid, 256, 0, stream>>>(W0, Wt0);
    transpose_to_bf16<<<tgrid, 256, 0, stream>>>(W1, Wt1);
    const int nx = T_STEPS * BATCH * HID;
    convert_f32_to_bf16<<<(nx + 255) / 256, 256, 0, stream>>>(X, Xbf, nx);

    const size_t step = (size_t)BATCH * HID;

    // Layer 0 (recurrent, sequential over T).
    for (int t = 0; t < T_STEPS; ++t) {
        lstm_step_kernel<<<16, 256, SMEM_BYTES, stream>>>(
            Xbf + (size_t)t * step,
            (t == 0) ? H0 : H0 + (size_t)(t - 1) * step,
            Wt0, b0, c0,
            H0 + (size_t)t * step, nullptr, (t == 0) ? 1 : 0);
    }
    // Layer 1: input is layer-0 hidden sequence; fp32 h goes straight to d_out.
    for (int t = 0; t < T_STEPS; ++t) {
        lstm_step_kernel<<<16, 256, SMEM_BYTES, stream>>>(
            H0 + (size_t)t * step,
            (t == 0) ? H1 : H1 + (size_t)(t - 1) * step,
            Wt1, b1, c1,
            H1 + (size_t)t * step, out + (size_t)t * step, (t == 0) ? 1 : 0);
    }
}
